// HierarchicalSoftmax_52501680226669
// MI455X (gfx1250) — matrix operand order repaired and checked by reading, not compiled, not run on hardware
//
#include <hip/hip_runtime.h>
#include <math.h>

#define HID   1024
#define NOCC  128

// d_ws layout (int32 units)
#define WS_CNT 0      // 128 counts
#define WS_OFF 128    // 129 offsets
#define WS_CUR 257    // 128 cursors
#define WS_IDX 385    // N sample indices

typedef __attribute__((ext_vector_type(16))) __bf16 v16bf;
typedef __attribute__((ext_vector_type(8)))  float  v8f;

union Frag { unsigned u[8]; v16bf bf; };            // trivially-copyable
struct __align__(8)  F2 { float x, y; };            // POD 8-byte load
struct __align__(16) U4 { unsigned a, b, c, d; };   // POD 16-byte (ds_load_b128)

__device__ __forceinline__ unsigned short f2bf(float f) {
    unsigned u = __float_as_uint(f);
    return (unsigned short)((u + 0x7FFFu + ((u >> 16) & 1u)) >> 16);
}
__device__ __forceinline__ float bf2f(unsigned short h) {
    return __uint_as_float(((unsigned)h) << 16);
}
__device__ __forceinline__ void split_bf(float x, unsigned short& h, unsigned short& l) {
    h = f2bf(x);
    l = f2bf(x - bf2f(h));
}

__device__ __forceinline__ float getW(int k, int col, int p, int o,
                                      const float* Wp, const float* W1, const float* W2) {
    if (col < 8)  return Wp[k * 8 + col];
    if (col < 24) return W1[((size_t)p * HID + k) * 16 + (col - 8)];
    if (col < 56) return W2[((size_t)o * HID + k) * 32 + (col - 24)];
    return 0.f;
}

__device__ __forceinline__ float softsel(const float* L, const float* bias,
                                         int off, int cnt, int sel) {
    float mx = -1e30f;
    for (int c = 0; c < cnt; ++c) mx = fmaxf(mx, L[off + c] + bias[off + c]);
    float s = 0.f, se = 0.f;
    for (int c = 0; c < cnt; ++c) {
        float e = __expf(L[off + c] + bias[off + c] - mx);
        s += e;
        if (c == sel) se = e;
    }
    return se / s;
}

// ---------------- bucketing kernels ----------------
__global__ void hs_zero(int* ws) {
    int i = threadIdx.x;
    if (i < 128) ws[WS_CNT + i] = 0;
}
__global__ void hs_count(const int* __restrict__ occ, int* ws, int n) {
    int i = blockIdx.x * blockDim.x + threadIdx.x;
    if (i < n) atomicAdd(&ws[WS_CNT + occ[i]], 1);
}
__global__ void hs_scan(int* ws) {
    if (threadIdx.x == 0 && blockIdx.x == 0) {
        int run = 0;
        for (int o = 0; o < NOCC; ++o) {
            int c = ws[WS_CNT + o];
            ws[WS_OFF + o] = run;
            ws[WS_CUR + o] = run;
            run += c;
        }
        ws[WS_OFF + NOCC] = run;
    }
}
__global__ void hs_fill(const int* __restrict__ occ, int* ws, int n) {
    int i = blockIdx.x * blockDim.x + threadIdx.x;
    if (i < n) {
        int pos = atomicAdd(&ws[WS_CUR + occ[i]], 1);
        ws[WS_IDX + pos] = i;
    }
}

// ---------------- main fused kernel ----------------
// One block per occurrence expert. Stage hi/lo bf16 weight panels (fragment
// order) in LDS once, then each wave GEMMs gathered 16-row X tiles against the
// 1024x64 fused panel with v_wmma_f32_16x16x32_bf16 (3-product hi/lo split),
// then does the 3 softmax-gathers per row.
__global__ void __launch_bounds__(128)
hs_main(const float* __restrict__ X,
        const int* __restrict__ Xph, const int* __restrict__ Xoc, const int* __restrict__ Xsb,
        const float* __restrict__ Wp, const float* __restrict__ bp,
        const float* __restrict__ W1, const float* __restrict__ b1,
        const float* __restrict__ W2, const float* __restrict__ b2,
        const int* __restrict__ ws, float* __restrict__ out, int n)
{
    extern __shared__ unsigned char smem[];
    unsigned* Bhi  = (unsigned*)smem;          // [32 ksteps][4 ntiles][32 lanes][8] u32 = 128KB
    unsigned* Blo  = Bhi + 32768;              // 128KB
    float*    bias = (float*)(Blo + 32768);    // 64 floats
    float*    tiles = bias + 64;               // 4 waves * 16 * 68 floats

    const int o = blockIdx.x;
    const int p = o >> 4;
    const int start = ws[WS_OFF + o];
    const int count = ws[WS_OFF + o + 1] - start;
    if (count == 0) return;
    const int tid = threadIdx.x;

    // fused bias row: [bp(8) | b1[p](16) | b2[o](32) | 0(8)]
    if (tid < 64) {
        float b = 0.f;
        if (tid < 8)       b = bp[tid];
        else if (tid < 24) b = b1[p * 16 + (tid - 8)];
        else if (tid < 56) b = b2[o * 32 + (tid - 24)];
        bias[tid] = b;
    }

    // stage B panels in WMMA fragment order:
    // frag element f = ((ks*4+nt)*32+lane)*8+v ; col = nt*16 + lane%16 ;
    // K = ks*32 + 16*(lane>=16) + 2v  (packed pair K, K+1)
    for (int f = tid; f < 32768; f += 128) {
        int v  = f & 7;
        int ln = (f >> 3) & 31;
        int nt = (f >> 8) & 3;
        int ks = f >> 10;
        int col = nt * 16 + (ln & 15);
        int k = ks * 32 + ((ln >> 4) << 4) + (v << 1);
        float w0 = getW(k,     col, p, o, Wp, W1, W2);
        float w1 = getW(k + 1, col, p, o, Wp, W1, W2);
        unsigned short h0, l0, h1, l1;
        split_bf(w0, h0, l0);
        split_bf(w1, h1, l1);
        Bhi[f] = (unsigned)h0 | ((unsigned)h1 << 16);
        Blo[f] = (unsigned)l0 | ((unsigned)l1 << 16);
    }
    __syncthreads();

    const int wave = tid >> 5;
    const int lane = tid & 31;
    const int* idxl = ws + WS_IDX + start;
    float* tl = tiles + wave * (16 * 68);
    const int klA = (lane >> 4) * 16;  // A-fragment K offset per lane half

    for (int tb = wave * 16; tb < count; tb += 64) {
        int rr = tb + (lane & 15);
        if (rr >= count) rr = count - 1;               // pad with a valid row
        const float* xr = X + (size_t)idxl[rr] * HID;

        // warm the stream: prefetch the first 1KB of this lane's row
        __builtin_prefetch(xr, 0, 3);
        __builtin_prefetch(xr + 128, 0, 3);

        v8f acc[4];
        const v8f vz = {0.f, 0.f, 0.f, 0.f, 0.f, 0.f, 0.f, 0.f};
        acc[0] = vz; acc[1] = vz; acc[2] = vz; acc[3] = vz;

        for (int ks = 0; ks < 32; ++ks) {
            // prefetch ~8 K-steps (1KB) ahead in this lane's row
            if (ks < 24) __builtin_prefetch(xr + (ks + 8) * 32 + klA, 0, 3);

            // build A hi/lo fragments (16x32 bf16): VGPR v holds K pair
            // 2v + 16*(lane>=16)
            Frag ah, al;
            #pragma unroll
            for (int v = 0; v < 8; ++v) {
                int k0 = ks * 32 + (v << 1) + klA;
                F2 xv = *(const F2*)(xr + k0);
                unsigned short h0, l0, h1, l1;
                split_bf(xv.x, h0, l0);
                split_bf(xv.y, h1, l1);
                ah.u[v] = (unsigned)h0 | ((unsigned)h1 << 16);
                al.u[v] = (unsigned)l0 | ((unsigned)l1 << 16);
            }
            #pragma unroll
            for (int nt = 0; nt < 4; ++nt) {
                int base = ((ks * 4 + nt) * 32 + lane) * 8;   // 32B-aligned
                Frag bh, bl;
                U4 q;
                q = *(const U4*)(Bhi + base);
                bh.u[0] = q.a; bh.u[1] = q.b; bh.u[2] = q.c; bh.u[3] = q.d;
                q = *(const U4*)(Bhi + base + 4);
                bh.u[4] = q.a; bh.u[5] = q.b; bh.u[6] = q.c; bh.u[7] = q.d;
                q = *(const U4*)(Blo + base);
                bl.u[0] = q.a; bl.u[1] = q.b; bl.u[2] = q.c; bl.u[3] = q.d;
                q = *(const U4*)(Blo + base + 4);
                bl.u[4] = q.a; bl.u[5] = q.b; bl.u[6] = q.c; bl.u[7] = q.d;
                // D = ah*bh + ah*bl + al*bh + C   (lo*lo dropped, ~2^-18 rel)
                acc[nt] = __builtin_amdgcn_wmma_f32_16x16x32_bf16(
                    false, al.bf, false, bh.bf, (short)0, acc[nt], false, false);
                acc[nt] = __builtin_amdgcn_wmma_f32_16x16x32_bf16(
                    false, ah.bf, false, bl.bf, (short)0, acc[nt], false, false);
                acc[nt] = __builtin_amdgcn_wmma_f32_16x16x32_bf16(
                    false, ah.bf, false, bh.bf, (short)0, acc[nt], false, false);
            }
        }

        // spill D tile to LDS: lane l, vgpr j -> (M = j + 8*(l>=16), N = l%16)
        #pragma unroll
        for (int j = 0; j < 8; ++j) {
            int m = j + ((lane >> 4) << 3);
            #pragma unroll
            for (int nt = 0; nt < 4; ++nt)
                tl[m * 68 + nt * 16 + (lane & 15)] = acc[nt][j];
        }
        __builtin_amdgcn_wave_barrier();   // same-wave DS ops are in-order; block reordering

        if (lane < 16) {
            int gr = tb + lane;
            if (gr < count) {
                int nidx = idxl[gr];
                const float* L = tl + lane * 68;
                float pp = softsel(L, bias, 0,  8,  Xph[nidx]);
                float po = pp * softsel(L, bias, 8,  16, Xoc[nidx] & 15);
                float ps = po * softsel(L, bias, 24, 32, Xsb[nidx] & 31);
                out[nidx]         = pp;
                out[n + nidx]     = po;
                out[2 * n + nidx] = ps;
            }
        }
        __builtin_amdgcn_wave_barrier();
    }
}

extern "C" void kernel_launch(void* const* d_in, const int* in_sizes, int n_in,
                              void* d_out, int out_size, void* d_ws, size_t ws_size,
                              hipStream_t stream) {
    const float* X   = (const float*)d_in[0];
    const int*   Xsb = (const int*)d_in[2];
    const int*   Xoc = (const int*)d_in[3];
    const int*   Xph = (const int*)d_in[4];
    const float* Wp  = (const float*)d_in[5];
    const float* bp  = (const float*)d_in[6];
    const float* W1  = (const float*)d_in[7];
    const float* b1  = (const float*)d_in[8];
    const float* W2  = (const float*)d_in[9];
    const float* b2  = (const float*)d_in[10];
    float* out = (float*)d_out;
    int n = in_sizes[1];                     // N (X_length has N elements)
    int* ws = (int*)d_ws;

    hs_zero<<<1, 128, 0, stream>>>(ws);
    int nb = (n + 255) / 256;
    hs_count<<<nb, 256, 0, stream>>>(Xoc, ws, n);
    hs_scan<<<1, 1, 0, stream>>>(ws);
    hs_fill<<<nb, 256, 0, stream>>>(Xoc, ws, n);

    size_t smem = (size_t)2 * 32768 * 4      // Bhi + Blo panels
                + 64 * 4                     // fused bias
                + 4 * 16 * 68 * 4;           // per-wave logit tiles
    hs_main<<<NOCC, 128, smem, stream>>>(X, Xph, Xoc, Xsb,
                                         Wp, bp, W1, b1, W2, b2,
                                         ws, out, n);
}
